// LSTM_55362128445749
// MI455X (gfx1250) — matrix operand order repaired and checked by reading, not compiled, not run on hardware
//
#include <hip/hip_runtime.h>
#include <cstdint>
#include <cstddef>

#define T_SEQ   512
#define B_BATCH 256
#define I_IN    512
#define H_HID   256
#define G4      1024   // 4*H
#define TC      32     // time-chunk length

typedef __attribute__((ext_vector_type(16))) __bf16 v16bf;
typedef __attribute__((ext_vector_type(8)))  float  v8f;
typedef __attribute__((ext_vector_type(4)))  int    v4i;

#ifndef __has_builtin
#define __has_builtin(x) 0
#endif
#if __has_builtin(__builtin_amdgcn_global_load_async_to_lds_b128) && \
    __has_builtin(__builtin_amdgcn_s_wait_asynccnt)
#define HAS_ASYNC_LDS 1
#else
#define HAS_ASYNC_LDS 0
#endif

// ---------------------------------------------------------------- helpers

// 16-byte global -> LDS copy. On gfx1250 with a capable toolchain this is a
// VGPR-bypassing async DMA (ASYNCcnt); otherwise a plain load/store pair.
__device__ __forceinline__ void cp_async16(void* lds, const void* gsrc) {
#if HAS_ASYNC_LDS
  __builtin_amdgcn_global_load_async_to_lds_b128((v4i*)gsrc, (v4i*)lds, 0, 0);
#else
  *(uint4*)lds = *(const uint4*)gsrc;
#endif
}

__device__ __forceinline__ void async_drain() {
#if HAS_ASYNC_LDS
  __builtin_amdgcn_s_wait_asynccnt(0);
#endif
}

__device__ __forceinline__ unsigned short f2bf(float f) {
  union { float f; unsigned u; } a; a.f = f;
  unsigned r = a.u + 0x7FFFu + ((a.u >> 16) & 1u);  // RNE
  return (unsigned short)(r >> 16);
}

__device__ __forceinline__ v16bf ld_frag(const unsigned short* p0,
                                         const unsigned short* p1) {
  union { uint4 u[2]; v16bf v; } t;
  t.u[0] = *(const uint4*)p0;
  t.u[1] = *(const uint4*)p1;
  return t.v;
}

__device__ __forceinline__ v8f wmma_bf16(v16bf a, v16bf b, v8f c) {
  // D = A(16x32 bf16) * B(32x16 bf16) + C(16x16 f32)
  return __builtin_amdgcn_wmma_f32_16x16x32_bf16(false, a, false, b,
                                                 (short)0, c, false, false);
}

__device__ __forceinline__ float sigmoidf_fast(float x) {
  return 1.0f / (1.0f + __expf(-x));
}
__device__ __forceinline__ float tanhf_fast(float x) {
  x = fminf(fmaxf(x, -20.0f), 20.0f);
  float e = __expf(-2.0f * x);
  return (1.0f - e) / (1.0f + e);
}

// ---------------------------------------------------------------- GEMM
// C[M,N] (f32, row-major, ldc) = A[M,K] * W[N,K]^T + bias[N]
// A is bf16 (row-major, lda, async-staged) or f32 (converted while staging).
// W is bf16, row-major (ldb) -- matches PyTorch (4H, d_in) weight layout.
// Double-buffered LDS; async stage of slab i+1 overlaps WMMAs of slab i.
// Requires M%128==0, N%128==0, K%32==0, grid = (N/128, M/128), block = 256.

__global__ __launch_bounds__(256) void gemm_bias_kernel(
    const void* __restrict__ Aptr, int a_is_f32, long lda,
    const unsigned short* __restrict__ Bw, long ldb,
    const float* __restrict__ bias,
    float* __restrict__ C, long ldc, int K)
{
  __shared__ unsigned short Al[2][128 * 32];  // [buf][row][k]  2x8KB
  __shared__ unsigned short Bl[2][128 * 32];  // [buf][col][k]  2x8KB

  const int tid  = threadIdx.x;
  const int lane = tid & 31;
  const int w    = tid >> 5;
  const int hi   = lane >> 4;
  const int ln   = lane & 15;
  const long m0  = (long)blockIdx.y * 128;
  const long n0  = (long)blockIdx.x * 128;
  const int  wm  = w & 1;    // 2 M-halves of 64
  const int  wn  = w >> 1;   // 4 N-quarters of 32

  v8f acc[4][2] = {};

  const int srow  = tid >> 1;
  const int shalf = tid & 1;

  auto stageA = [&](int kt, int buf) {
    if (a_is_f32) {
      const float* g = (const float*)Aptr + (m0 + srow) * lda + kt + shalf * 16;
      union { unsigned short s[16]; uint4 u[2]; } t;
      #pragma unroll
      for (int i = 0; i < 16; ++i) t.s[i] = f2bf(g[i]);
      uint4* d = (uint4*)&Al[buf][srow * 32 + shalf * 16];
      d[0] = t.u[0]; d[1] = t.u[1];
    } else {
      const unsigned short* g =
          (const unsigned short*)Aptr + (m0 + srow) * lda + kt + shalf * 16;
      unsigned short* d = &Al[buf][srow * 32 + shalf * 16];
      cp_async16(d, g);
      cp_async16(d + 8, g + 8);
    }
  };
  auto stageB = [&](int kt, int buf) {
    const unsigned short* g = Bw + (n0 + srow) * ldb + kt + shalf * 16;
    unsigned short* d = &Bl[buf][srow * 32 + shalf * 16];
    cp_async16(d, g);
    cp_async16(d + 8, g + 8);
  };

  const int nk = K >> 5;
  stageA(0, 0);
  stageB(0, 0);

  for (int i = 0; i < nk; ++i) {
    const int cb = i & 1;
    async_drain();       // my slab-i async copies done
    __syncthreads();     // everyone's slab-i copies visible
    if (i + 1 < nk) {    // async prefetch of slab i+1 overlaps WMMAs below
      stageA((i + 1) << 5, cb ^ 1);
      stageB((i + 1) << 5, cb ^ 1);
    }
    const unsigned short* As = Al[cb];
    const unsigned short* Bs = Bl[cb];
    #pragma unroll
    for (int ni = 0; ni < 2; ++ni) {
      const int bc = wn * 32 + ni * 16 + ln;
      // B frag: lane=col, e<8 -> K = e + hi*8 ; e>=8 -> K = 16 + (e-8) + hi*8
      v16bf bf = ld_frag(&Bs[bc * 32 + hi * 8], &Bs[bc * 32 + hi * 8 + 16]);
      #pragma unroll
      for (int mi = 0; mi < 4; ++mi) {
        const int ar = wm * 64 + mi * 16 + ln;
        // A frag: lane=row, e<8 -> K = e + hi*8 ; e>=8 -> K = 16 + (e-8) + hi*8
        v16bf af = ld_frag(&As[ar * 32 + hi * 8], &As[ar * 32 + hi * 8 + 16]);
        acc[mi][ni] = wmma_bf16(af, bf, acc[mi][ni]);
      }
    }
  }

  // ---- epilogue: bias + store (D layout: VGPR r -> M = r + 8*hi, N = lane&15)
  #pragma unroll
  for (int ni = 0; ni < 2; ++ni) {
    const long col = n0 + wn * 32 + ni * 16 + ln;
    const float bv = bias ? bias[col] : 0.0f;
    #pragma unroll
    for (int mi = 0; mi < 4; ++mi) {
      #pragma unroll
      for (int r = 0; r < 8; ++r) {
        const long row = m0 + wm * 64 + mi * 16 + r + 8 * hi;
        C[row * ldc + col] = acc[mi][ni][r] + bv;
      }
    }
  }
}

// ---------------------------------------------------------------- LSTM scan
// Persistent 64 WGs (4 batch-tiles x 16 hidden-tiles). Each WG:
//   - async-stages its 64 whh columns (4 gates x 16 hidden) to LDS ONCE
//   - keeps its c tile (64x16) in registers across all steps
//   - per step: async-stage h slice, GEMM g += h @ whh^T (bf16 WMMA),
//     gates, write h (ping-pong), one atomic-counter grid barrier

__global__ __launch_bounds__(256) void lstm_scan_kernel(
    const float* __restrict__ xg,          // [TC][B][4H] f32 (chunk-local)
    const unsigned short* __restrict__ whh,// [4H][H] bf16
    unsigned short* __restrict__ hbuf,     // 2 x [B][H] bf16 ping-pong
    float* __restrict__ cstate,            // [B][H] f32
    unsigned short* __restrict__ ys,       // [T][B][H] bf16 (layer output)
    unsigned int* __restrict__ bar,        // zeroed before launch
    int t0, int nsteps)
{
  __shared__ __align__(16) unsigned char smem[64 * 1024];
  unsigned short* Bl = (unsigned short*)smem;                // [64 col][256 k] 32KB
  unsigned short* Al = (unsigned short*)(smem + 32 * 1024);  // [64 row][256 k] 32KB
  float*          Gl = (float*)(smem + 32 * 1024);           // [64 row][64 col] aliases Al

  const int tid  = threadIdx.x;
  const int lane = tid & 31, w = tid >> 5;
  const int hi   = lane >> 4, ln = lane & 15;
  const int wg   = blockIdx.x;
  const int m0   = (wg >> 4) * 64;   // batch tile
  const int n0   = (wg & 15) * 16;   // hidden tile
  const int wm   = w & 1;            // 2 M-halves of 32
  const int wn   = w >> 1;           // wave's gate (N-tile of 16)

  // ---- async-stage whh tile once: col j -> gate=j>>4, hidden=n0+(j&15)
  {
    const int col  = tid >> 2;
    const int part = tid & 3;
    const long wrow = (long)(col >> 4) * H_HID + n0 + (col & 15);
    const unsigned short* g = whh + wrow * H_HID + part * 64;
    unsigned short* d = &Bl[col * 256 + part * 64];
    #pragma unroll
    for (int i = 0; i < 8; ++i) cp_async16(d + i * 8, g + i * 8);
  }

  // ---- per-thread cell-state: row um = tid>>2, 4 hidden at uh=(tid&3)*4
  const int um = tid >> 2;
  const int uh = (tid & 3) * 4;
  float4 c4 = *(const float4*)&cstate[(m0 + um) * H_HID + n0 + uh];

  const unsigned nwg = gridDim.x;

  for (int st = 0; st < nsteps; ++st) {
    const int s = t0 + st;
    const unsigned short* hr = hbuf + (size_t)(s & 1) * (B_BATCH * H_HID);
    unsigned short*       hw = hbuf + (size_t)((s + 1) & 1) * (B_BATCH * H_HID);

    __syncthreads();   // previous step's Gl reads done before Al overwrite
    {  // async-stage h slice (64 x 256 bf16)
      const int row = tid >> 2, part = tid & 3;
      const unsigned short* g = hr + (size_t)(m0 + row) * H_HID + part * 64;
      unsigned short* d = &Al[row * 256 + part * 64];
      #pragma unroll
      for (int i = 0; i < 8; ++i) cp_async16(d + i * 8, g + i * 8);
    }
    async_drain();     // h slice (and, on step 0, whh) landed in LDS
    __syncthreads();

    // ---- GEMM 64x64x256: 16 WMMAs per wave
    v8f acc[2] = {};
    #pragma unroll
    for (int kt = 0; kt < 8; ++kt) {
      const int bc = wn * 16 + ln;
      v16bf bf = ld_frag(&Bl[bc * 256 + kt * 32 + hi * 8],
                         &Bl[bc * 256 + kt * 32 + hi * 8 + 16]);
      #pragma unroll
      for (int mi = 0; mi < 2; ++mi) {
        const int ar = wm * 32 + mi * 16 + ln;
        v16bf af = ld_frag(&Al[ar * 256 + kt * 32 + hi * 8],
                           &Al[ar * 256 + kt * 32 + hi * 8 + 16]);
        acc[mi] = wmma_bf16(af, bf, acc[mi]);
      }
    }
    __syncthreads();   // all Al reads done -> safe to alias with Gl
    #pragma unroll
    for (int mi = 0; mi < 2; ++mi)
      #pragma unroll
      for (int r = 0; r < 8; ++r)
        Gl[(wm * 32 + mi * 16 + r + 8 * hi) * 64 + wn * 16 + ln] = acc[mi][r];
    __syncthreads();

    // ---- gates + state update (per-thread 4 elems, gate order i,f,g,o)
    {
      const float* xr = xg + ((size_t)st * B_BATCH + (m0 + um)) * G4 + n0 + uh;
      const float* gr = Gl + um * 64 + uh;
      float cc[4] = {c4.x, c4.y, c4.z, c4.w};
      union { unsigned short sh[4]; uint2 u; } hb;
      #pragma unroll
      for (int k = 0; k < 4; ++k) {
        const float gi = xr[0 * H_HID + k] + gr[ 0 + k];
        const float gf = xr[1 * H_HID + k] + gr[16 + k];
        const float gg = xr[2 * H_HID + k] + gr[32 + k];
        const float go = xr[3 * H_HID + k] + gr[48 + k];
        const float cn = sigmoidf_fast(gf) * cc[k] +
                         sigmoidf_fast(gi) * tanhf_fast(gg);
        cc[k] = cn;
        hb.sh[k] = f2bf(sigmoidf_fast(go) * tanhf_fast(cn));
      }
      c4 = make_float4(cc[0], cc[1], cc[2], cc[3]);
      *(uint2*)&hw[(size_t)(m0 + um) * H_HID + n0 + uh] = hb.u;
      *(uint2*)&ys[((size_t)s * B_BATCH + (m0 + um)) * H_HID + n0 + uh] = hb.u;
    }

    // ---- grid barrier (counter monotonically increases: target = nwg*(st+1))
    __syncthreads();
    if (tid == 0) {
      __threadfence();
      atomicAdd(bar, 1u);
      const unsigned target = nwg * (unsigned)(st + 1);
      while (atomicAdd(bar, 0u) < target) __builtin_amdgcn_s_sleep(2);
    }
    __syncthreads();
  }

  *(float4*)&cstate[(m0 + um) * H_HID + n0 + uh] = c4;
}

// ---------------------------------------------------------------- tiny utils

__global__ void f32_to_bf16_kernel(const float* __restrict__ s,
                                   unsigned short* __restrict__ d, int n) {
  int i = blockIdx.x * blockDim.x + threadIdx.x;
  if (i < n) d[i] = f2bf(s[i]);
}
__global__ void vec_add_kernel(const float* __restrict__ a,
                               const float* __restrict__ b,
                               float* __restrict__ d, int n) {
  int i = blockIdx.x * blockDim.x + threadIdx.x;
  if (i < n) d[i] = a[i] + b[i];
}
__global__ void zero_u32_kernel(unsigned int* p, int n) {
  int i = blockIdx.x * blockDim.x + threadIdx.x;
  if (i < n) p[i] = 0u;
}

// ---------------------------------------------------------------- driver

extern "C" void kernel_launch(void* const* d_in, const int* in_sizes, int n_in,
                              void* d_out, int out_size, void* d_ws, size_t ws_size,
                              hipStream_t stream)
{
  (void)in_sizes; (void)n_in; (void)out_size; (void)ws_size;

  const float* x = (const float*)d_in[0];
  const float *wihf[4], *whhf[4], *bihf[4], *bhhf[4];
  for (int l = 0; l < 4; ++l) {
    wihf[l] = (const float*)d_in[1 + 4 * l];
    whhf[l] = (const float*)d_in[2 + 4 * l];
    bihf[l] = (const float*)d_in[3 + 4 * l];
    bhhf[l] = (const float*)d_in[4 + 4 * l];
  }
  const float* lrw = (const float*)d_in[17];
  const float* lrb = (const float*)d_in[18];

  const int din[4] = {I_IN, H_HID, H_HID, H_HID};

  char* ws = (char*)d_ws;
  size_t off = 0;
  auto alloc = [&](size_t bytes) -> char* {
    char* p = ws + off;
    off = (off + bytes + 255) & ~(size_t)255;
    return p;
  };

  unsigned short* whh_bf[4]; unsigned short* wih_bf[4]; float* bsum[4];
  for (int l = 0; l < 4; ++l) whh_bf[l] = (unsigned short*)alloc((size_t)G4 * H_HID * 2);
  for (int l = 0; l < 4; ++l) wih_bf[l] = (unsigned short*)alloc((size_t)G4 * din[l] * 2);
  for (int l = 0; l < 4; ++l) bsum[l]   = (float*)alloc((size_t)G4 * 4);
  unsigned short* lrw_bf = (unsigned short*)alloc((size_t)H_HID * H_HID * 2);
  float*          xg     = (float*)alloc((size_t)TC * B_BATCH * G4 * 4);
  unsigned short* ysA    = (unsigned short*)alloc((size_t)T_SEQ * B_BATCH * H_HID * 2);
  unsigned short* ysB    = (unsigned short*)alloc((size_t)T_SEQ * B_BATCH * H_HID * 2);
  unsigned short* hping  = (unsigned short*)alloc((size_t)2 * B_BATCH * H_HID * 2);
  float*          cst    = (float*)alloc((size_t)B_BATCH * H_HID * 4);
  unsigned int*   bar    = (unsigned int*)alloc(256);

  // ---- one-time weight conversion + bias pre-sum
  for (int l = 0; l < 4; ++l) {
    const int nw = G4 * din[l];
    f32_to_bf16_kernel<<<(nw + 255) / 256, 256, 0, stream>>>(wihf[l], wih_bf[l], nw);
    const int nh = G4 * H_HID;
    f32_to_bf16_kernel<<<(nh + 255) / 256, 256, 0, stream>>>(whhf[l], whh_bf[l], nh);
    vec_add_kernel<<<(G4 + 255) / 256, 256, 0, stream>>>(bihf[l], bhhf[l], bsum[l], G4);
  }
  {
    const int nl = H_HID * H_HID;
    f32_to_bf16_kernel<<<(nl + 255) / 256, 256, 0, stream>>>(lrw, lrw_bf, nl);
  }

  // ---- layers
  const void* curIn = (const void*)x;
  int curF32 = 1;
  unsigned short* ybufs[2] = {ysA, ysB};
  unsigned short* yout = ysB;

  for (int l = 0; l < 4; ++l) {
    yout = ybufs[l & 1];
    zero_u32_kernel<<<(B_BATCH * H_HID + 255) / 256, 256, 0, stream>>>(
        (unsigned int*)hping, B_BATCH * H_HID);   // 2*B*H bf16 == B*H u32
    zero_u32_kernel<<<(B_BATCH * H_HID + 255) / 256, 256, 0, stream>>>(
        (unsigned int*)cst, B_BATCH * H_HID);

    const long lda = din[l];
    for (int ch = 0; ch < T_SEQ / TC; ++ch) {
      const void* aChunk = curF32
        ? (const void*)((const float*)curIn + (size_t)ch * TC * B_BATCH * lda)
        : (const void*)((const unsigned short*)curIn + (size_t)ch * TC * B_BATCH * lda);
      dim3 gg(G4 / 128, (TC * B_BATCH) / 128);
      gemm_bias_kernel<<<gg, 256, 0, stream>>>(aChunk, curF32, lda,
                                               wih_bf[l], lda, bsum[l],
                                               xg, (long)G4, din[l]);
      zero_u32_kernel<<<1, 32, 0, stream>>>(bar, 1);
      lstm_scan_kernel<<<64, 256, 0, stream>>>(xg, whh_bf[l], hping, cst,
                                               yout, bar, ch * TC, TC);
    }
    curIn = (const void*)yout;
    curF32 = 0;
  }

  // ---- final projection: rows ys[t, B-1, :] -> d_out (T x H, f32)
  {
    const unsigned short* Afin = yout + (size_t)(B_BATCH - 1) * H_HID;
    dim3 gg(H_HID / 128, T_SEQ / 128);
    gemm_bias_kernel<<<gg, 256, 0, stream>>>((const void*)Afin, 0,
                                             (long)B_BATCH * H_HID,
                                             lrw_bf, (long)H_HID, lrb,
                                             (float*)d_out, (long)H_HID, H_HID);
  }
}
